// DeformableTransformer_9079560864322
// MI455X (gfx1250) — matrix operand order repaired and checked by reading, not compiled, not run on hardware
//
#include <hip/hip_runtime.h>
#include <cstdint>
#include <cstddef>

// ---------------------------------------------------------------------------
// Types for CDNA5 WMMA (gfx1250, wave32)
// ---------------------------------------------------------------------------
typedef __bf16 bf16_t;
typedef __attribute__((ext_vector_type(16))) __bf16 v16bf;
typedef __attribute__((ext_vector_type(8)))  __bf16 v8bf;
typedef __attribute__((ext_vector_type(8)))  float  v8f;

#define NQ      63180      // total queries (sum of level sizes)
#define CDIM    256
#define NHEADS  8
#define NLVL    4
#define NPTS    4
#define DFFDIM  1024
#define NLAYER  6

// level boundaries: sizes 55296, 6912, 864, 108 ; starts 0, 55296, 62208, 63072
static __device__ __forceinline__ int level_of(int n, int& s) {
  if (n < 55296) { s = n;          return 0; }
  if (n < 62208) { s = n - 55296;  return 1; }
  if (n < 63072) { s = n - 62208;  return 2; }
  s = n - 63072; return 3;
}

static __device__ __forceinline__ void wait_asynccnt_2() {
#if __has_builtin(__builtin_amdgcn_s_wait_asynccnt)
  __builtin_amdgcn_s_wait_asynccnt(2);
#else
  asm volatile("s_wait_asynccnt 2" ::: "memory");
#endif
}
static __device__ __forceinline__ void wait_asynccnt_0() {
#if __has_builtin(__builtin_amdgcn_s_wait_asynccnt)
  __builtin_amdgcn_s_wait_asynccnt(0);
#else
  asm volatile("s_wait_asynccnt 0" ::: "memory");
#endif
}

// ---------------------------------------------------------------------------
// Weight pre-pack: W [K x Nout] f32 row-major -> per-lane B fragments (bf16).
// B-matrix lane layout (ISA 7.12.2): lane = 16*half + n16 ;
//   e<8 : K = kk*32 + 8*half + e ; e>=8 : K = kk*32 + 8*half + e + 8
// Packed address: ((nTile*kTiles + kk)*32 + lane)*16  (16 bf16 per lane slot)
// ---------------------------------------------------------------------------
__global__ void pack_w_kernel(const float* __restrict__ W, bf16_t* __restrict__ out,
                              int K, int Nout) {
  int tid = blockIdx.x * blockDim.x + threadIdx.x;
  int kTiles = K >> 5;
  int total = (Nout >> 4) * kTiles * 32;
  if (tid >= total) return;
  int lane  = tid & 31;
  int rest  = tid >> 5;
  int kk    = rest % kTiles;
  int nTile = rest / kTiles;
  int hlf   = lane >> 4;
  int n     = (nTile << 4) + (lane & 15);
  union { bf16_t s[16]; v16bf v; } frag;
#pragma unroll
  for (int e = 0; e < 16; ++e) {
    int k = (kk << 5) + (hlf << 3) + (e < 8 ? e : e + 8);
    frag.s[e] = (bf16_t)W[(size_t)k * Nout + n];
  }
  *(v16bf*)(out + (size_t)tid * 16) = frag.v;
}

// ---------------------------------------------------------------------------
// WMMA GEMM with async-LDS double-buffered A staging.
//   out[M x Nout] = A[M x K](bf16) * Wpk + bias   (optional relu)
// Block = 128 threads = 4 waves; block tile = 64 rows x 64 cols.
// A tile per k-step: 64 rows x 32 bf16 (64B/row), staged into LDS with
// global_load_async_to_lds_b128 (2 chunks/thread), 80B row pitch so the
// per-lane fragment ds_load_b128s are bank-conflict free.
// A-fragment layout (ISA 7.12.2): lane = 16*half + m ;
//   e<8 : K = kk*32 + 8*half + e ; e>=8 : K = kk*32 + 8*half + e + 8
//   -> per-lane bytes [16h,16h+16) and [32+16h, 48+16h) of the 64B row.
// EXEC is uniform at every WMMA; edge rows load clamped garbage whose D rows
// are simply never stored.
// ---------------------------------------------------------------------------
#define APITCH 80
template<typename TO>
__global__ void gemm_wmma_kernel(const bf16_t* __restrict__ A, const bf16_t* __restrict__ Wpk,
                                 const float* __restrict__ bias, TO* __restrict__ out,
                                 int M, int K, int Nout, int relu) {
  __shared__ __align__(16) unsigned char atile[2][64 * APITCH];
  const int tid     = threadIdx.x;
  const int lane    = tid & 31;
  const int wave    = tid >> 5;               // 0..3
  const int hlf     = lane >> 4;
  const int n16     = lane & 15;
  const int rowBase = blockIdx.x * 64 + wave * 16;
  const int nTile0  = blockIdx.y * 4;
  const int kTiles  = K >> 5;

  // --- staging: this thread's 2 chunks of the 64x64B A tile ---------------
  const int ch0 = tid, ch1 = tid + 128;
  const int r0 = ch0 >> 2, c0 = ch0 & 3;
  const int r1 = ch1 >> 2, c1 = ch1 & 3;
  int gr0 = blockIdx.x * 64 + r0; if (gr0 >= M) gr0 = M - 1;
  int gr1 = blockIdx.x * 64 + r1; if (gr1 >= M) gr1 = M - 1;
  const unsigned char* g0 = (const unsigned char*)(A + (size_t)gr0 * K) + c0 * 16;
  const unsigned char* g1 = (const unsigned char*)(A + (size_t)gr1 * K) + c1 * 16;
  const unsigned l0 = (unsigned)(size_t)&atile[0][r0 * APITCH + c0 * 16];
  const unsigned l1 = (unsigned)(size_t)&atile[0][r1 * APITCH + c1 * 16];
  const unsigned bufStride = 64 * APITCH;

  auto issue = [&](int kk) {
    unsigned b = (kk & 1) ? bufStride : 0u;
    unsigned long long ga0 = (unsigned long long)(size_t)(g0 + (size_t)kk * 64);
    unsigned long long ga1 = (unsigned long long)(size_t)(g1 + (size_t)kk * 64);
    unsigned la0 = l0 + b, la1 = l1 + b;
    asm volatile("global_load_async_to_lds_b128 %0, %1, off"
                 :: "v"(la0), "v"(ga0) : "memory");
    asm volatile("global_load_async_to_lds_b128 %0, %1, off"
                 :: "v"(la1), "v"(ga1) : "memory");
  };
  issue(0);
  if (kTiles > 1) issue(1);

  const int rL = wave * 16 + n16;   // this lane's A row within the tile

  v8f acc[4] = {};
  for (int kk = 0; kk < kTiles; ++kk) {
    // ensure tile kk has landed (tile kk+1's 2 async loads may be in flight)
    if (kk + 1 < kTiles) wait_asynccnt_2(); else wait_asynccnt_0();
    __syncthreads();

    // A fragment from LDS (2x ds_load_b128, conflict-free via 80B pitch)
    union { v8bf h[2]; v16bf v; } af;
    const unsigned char* tp = &atile[kk & 1][rL * APITCH + hlf * 16];
    af.h[0] = *(const v8bf*)(tp);
    af.h[1] = *(const v8bf*)(tp + 32);

    if (kk + 1 < kTiles) {
      __builtin_prefetch((const void*)(Wpk +
          ((((size_t)nTile0 * kTiles + (kk + 1)) * 32 + lane) * 16)), 0, 1);
    }
#pragma unroll
    for (int t = 0; t < 4; ++t) {
      const v16bf b = *(const v16bf*)(Wpk +
          ((((size_t)(nTile0 + t) * kTiles + kk) * 32 + lane) * 16));
      acc[t] = __builtin_amdgcn_wmma_f32_16x16x32_bf16(
          false, af.v, false, b, (short)0, acc[t], false, false);
    }

    __syncthreads();                       // everyone done reading tile kk
    if (kk + 2 < kTiles) issue(kk + 2);    // refill the buffer just freed
  }

  // D layout: vgpr j, lane = 16*half + n16  ->  (m = j + 8*half, n = n16)
#pragma unroll
  for (int t = 0; t < 4; ++t) {
    int col  = (nTile0 + t) * 16 + n16;
    float bb = bias ? bias[col] : 0.0f;
#pragma unroll
    for (int j = 0; j < 8; ++j) {
      int r = rowBase + j + 8 * hlf;
      if (r < M) {
        float v = acc[t][j] + bb;
        if (relu) v = v > 0.0f ? v : 0.0f;
        out[(size_t)r * Nout + col] = (TO)v;
      }
    }
  }
}

// ---------------------------------------------------------------------------
// Input flatten/transpose: src(B,C,D,H,W) per level -> x[n][c] (f32 + bf16)
// ---------------------------------------------------------------------------
__global__ void pack_src_kernel(const float* s0, const float* s1, const float* s2, const float* s3,
                                const float* p0, const float* p1, const float* p2, const float* p3,
                                const unsigned char* m0, const unsigned char* m1,
                                const unsigned char* m2, const unsigned char* m3,
                                const float* __restrict__ level_embed,
                                float* __restrict__ xflat, bf16_t* __restrict__ xb,
                                float* __restrict__ posflat,
                                unsigned char* __restrict__ maskflat) {
  int tid = blockIdx.x * blockDim.x + threadIdx.x;
  if (tid >= NQ * CDIM) return;
  int n = tid >> 8, c = tid & 255;
  int s, l = level_of(n, s);
  const float* sp; const float* pp; const unsigned char* mp; int size;
  switch (l) {
    case 0: sp = s0; pp = p0; mp = m0; size = 55296; break;
    case 1: sp = s1; pp = p1; mp = m1; size = 6912;  break;
    case 2: sp = s2; pp = p2; mp = m2; size = 864;   break;
    default:sp = s3; pp = p3; mp = m3; size = 108;   break;
  }
  float xv = sp[(size_t)c * size + s];
  xflat[tid] = xv;
  xb[tid]    = (bf16_t)xv;
  posflat[tid] = pp[(size_t)c * size + s] + level_embed[l * CDIM + c];
  if (c == 0) maskflat[n] = mp[s];
}

// valid ratios per level: [vw, vh, vd]
__global__ void valid_ratio_kernel(const unsigned char* m0, const unsigned char* m1,
                                   const unsigned char* m2, const unsigned char* m3,
                                   float* __restrict__ vr) {
  if (threadIdx.x != 0 || blockIdx.x != 0) return;
  const unsigned char* ms[4] = {m0, m1, m2, m3};
  const int D[4] = {24,12,6,3}, H[4] = {48,24,12,6}, W[4] = {48,24,12,6};
  for (int l = 0; l < 4; ++l) {
    int cd = 0, ch = 0, cw = 0;
    for (int z = 0; z < D[l]; ++z) cd += ms[l][(size_t)z * H[l] * W[l]] ? 0 : 1;
    for (int y = 0; y < H[l]; ++y) ch += ms[l][(size_t)y * W[l]] ? 0 : 1;
    for (int x = 0; x < W[l]; ++x) cw += ms[l][x] ? 0 : 1;
    vr[l*3+0] = (float)cw / W[l];
    vr[l*3+1] = (float)ch / H[l];
    vr[l*3+2] = (float)cd / D[l];
  }
}

// reference points: ref[n][l2][xyz] = base(n) * vr[l2]
__global__ void ref_points_kernel(const float* __restrict__ vr, float* __restrict__ ref) {
  int t = blockIdx.x * blockDim.x + threadIdx.x;
  if (t >= NQ * NLVL) return;
  int n = t >> 2, l2 = t & 3;
  int s, l = level_of(n, s);
  const int Ha[4] = {48,24,12,6}, Wd[4] = {48,24,12,6};
  const int Da[4] = {24,12,6,3};
  int Dl = Da[l], Hl = Ha[l], Wl = Wd[l];
  int xw = s % Wl, yh = (s / Wl) % Hl, zd = s / (Wl * Hl);
  float bx = (xw + 0.5f) / Wl, by = (yh + 0.5f) / Hl, bz = (zd + 0.5f) / Dl;
  ref[(size_t)t*3+0] = bx * vr[l2*3+0];
  ref[(size_t)t*3+1] = by * vr[l2*3+1];
  ref[(size_t)t*3+2] = bz * vr[l2*3+2];
}

// q = x + pos, emitted straight as bf16 (GEMM A operand only)
__global__ void add_bf16_kernel(const float* __restrict__ a, const float* __restrict__ b,
                                bf16_t* __restrict__ o, int n) {
  int t = blockIdx.x * blockDim.x + threadIdx.x;
  if (t < n) o[t] = (bf16_t)(a[t] + b[t]);
}

__global__ void mask_v_kernel(float* __restrict__ v, const unsigned char* __restrict__ mask) {
  int t = blockIdx.x * blockDim.x + threadIdx.x;
  if (t >= NQ * CDIM) return;
  if (mask[t >> 8]) v[t] = 0.0f;
}

// softmax over 16 contiguous logits per (n, head)
__global__ void softmax16_kernel(float* __restrict__ att, int total) {
  int t = blockIdx.x * blockDim.x + threadIdx.x;
  if (t >= total) return;
  float* p = att + (size_t)t * 16;
  float m = p[0];
#pragma unroll
  for (int i = 1; i < 16; ++i) m = fmaxf(m, p[i]);
  float e[16]; float s = 0.0f;
#pragma unroll
  for (int i = 0; i < 16; ++i) { e[i] = __expf(p[i] - m); s += e[i]; }
  float inv = 1.0f / s;
#pragma unroll
  for (int i = 0; i < 16; ++i) p[i] = e[i] * inv;
}

// ---------------------------------------------------------------------------
// MSDA trilinear sampling: one wave32 per (query, head); lane = head channel.
// Sampling math is wave-uniform; 32 lanes do coalesced 128B gathers.
// Output written as bf16 (feeds the Wo WMMA GEMM directly).
// ---------------------------------------------------------------------------
__global__ void msda_kernel(const float* __restrict__ v, const float* __restrict__ off,
                            const float* __restrict__ att, const float* __restrict__ ref,
                            bf16_t* __restrict__ outb) {
  int pair = blockIdx.x * (blockDim.x >> 5) + (threadIdx.x >> 5);
  int lane = threadIdx.x & 31;
  if (pair >= NQ * NHEADS) return;
  int n = pair / NHEADS, h = pair % NHEADS;
  const int Da[4] = {24,12,6,3}, Ha[4] = {48,24,12,6}, Wd[4] = {48,24,12,6};
  const int St[4] = {0, 55296, 62208, 63072};
  float acc = 0.0f;
#pragma unroll
  for (int l = 0; l < NLVL; ++l) {
    const int Dl = Da[l], Hl = Ha[l], Wl = Wd[l], start = St[l];
    const float rx = ref[((size_t)n * 4 + l) * 3 + 0];
    const float ry = ref[((size_t)n * 4 + l) * 3 + 1];
    const float rz = ref[((size_t)n * 4 + l) * 3 + 2];
#pragma unroll
    for (int p = 0; p < NPTS; ++p) {
      size_t oidx = ((((size_t)n * NHEADS + h) * NLVL + l) * NPTS + p) * 3;
      float px = rx * Wl + off[oidx + 0] - 0.5f;
      float py = ry * Hl + off[oidx + 1] - 0.5f;
      float pz = rz * Dl + off[oidx + 2] - 0.5f;
      float aw = att[((size_t)n * NHEADS + h) * 16 + l * NPTS + p];
      float x0 = floorf(px), y0 = floorf(py), z0 = floorf(pz);
#pragma unroll
      for (int corner = 0; corner < 8; ++corner) {
        float xi = x0 + (float)(corner & 1);
        float yi = y0 + (float)((corner >> 1) & 1);
        float zi = z0 + (float)((corner >> 2) & 1);
        bool ok = (xi >= 0.0f) && (xi < (float)Wl) &&
                  (yi >= 0.0f) && (yi < (float)Hl) &&
                  (zi >= 0.0f) && (zi < (float)Dl);
        if (ok) {
          float w = (1.0f - fabsf(px - xi)) * (1.0f - fabsf(py - yi)) * (1.0f - fabsf(pz - zi));
          int xc = (int)xi, yc = (int)yi, zc = (int)zi;
          int idx = (zc * Hl + yc) * Wl + xc;
          acc += w * aw * v[(size_t)(start + idx) * CDIM + h * 32 + lane];
        }
      }
    }
  }
  outb[(size_t)n * CDIM + h * 32 + lane] = (bf16_t)acc;
}

// residual + LayerNorm: one wave32 per row; writes f32 x and bf16 mirror
__global__ void ln_res_kernel(const float* __restrict__ x, const float* __restrict__ a,
                              const float* __restrict__ g, const float* __restrict__ b,
                              float* __restrict__ o, bf16_t* __restrict__ ob) {
  int row  = blockIdx.x * (blockDim.x >> 5) + (threadIdx.x >> 5);
  int lane = threadIdx.x & 31;
  if (row >= NQ) return;
  const float* xp = x + (size_t)row * CDIM;
  const float* ap = a + (size_t)row * CDIM;
  float vals[8]; float s = 0.0f;
#pragma unroll
  for (int i = 0; i < 8; ++i) {
    int c = lane + i * 32;
    vals[i] = xp[c] + ap[c];
    s += vals[i];
  }
#pragma unroll
  for (int off = 16; off > 0; off >>= 1) s += __shfl_xor(s, off, 32);
  float mean = s * (1.0f / CDIM);
  float vs = 0.0f;
#pragma unroll
  for (int i = 0; i < 8; ++i) { float d = vals[i] - mean; vs += d * d; }
#pragma unroll
  for (int off = 16; off > 0; off >>= 1) vs += __shfl_xor(vs, off, 32);
  float inv = rsqrtf(vs * (1.0f / CDIM) + 1e-5f);
#pragma unroll
  for (int i = 0; i < 8; ++i) {
    int c = lane + i * 32;
    float r = (vals[i] - mean) * inv * g[c] + b[c];
    o[(size_t)row * CDIM + c]  = r;
    ob[(size_t)row * CDIM + c] = (bf16_t)r;
  }
}

// final unflatten: x[n][c] -> per-level (B,C,D,H,W) fmaps concatenated flat
__global__ void unpack_kernel(const float* __restrict__ x, float* __restrict__ out) {
  int tid = blockIdx.x * blockDim.x + threadIdx.x;
  if (tid >= NQ * CDIM) return;
  int n = tid >> 8, c = tid & 255;
  int s, l = level_of(n, s);
  const int St[4] = {0, 55296, 62208, 63072};
  const int Sz[4] = {55296, 6912, 864, 108};
  out[(size_t)CDIM * St[l] + (size_t)c * Sz[l] + s] = x[tid];
}

// ---------------------------------------------------------------------------
// Host orchestration
// ---------------------------------------------------------------------------
extern "C" void kernel_launch(void* const* d_in, const int* in_sizes, int n_in,
                              void* d_out, int out_size, void* d_ws, size_t ws_size,
                              hipStream_t stream) {
  (void)in_sizes; (void)n_in; (void)out_size; (void)ws_size;
  // dict order: (src,pos,mask) x4 levels, then params
  const float* src[4] = {(const float*)d_in[0], (const float*)d_in[3],
                         (const float*)d_in[6], (const float*)d_in[9]};
  const float* pos[4] = {(const float*)d_in[1], (const float*)d_in[4],
                         (const float*)d_in[7], (const float*)d_in[10]};
  const unsigned char* msk[4] = {(const unsigned char*)d_in[2], (const unsigned char*)d_in[5],
                                 (const unsigned char*)d_in[8], (const unsigned char*)d_in[11]};
  const float* level_embed = (const float*)d_in[12];
  const float* Wso = (const float*)d_in[13]; const float* bso = (const float*)d_in[14];
  const float* Wat = (const float*)d_in[15]; const float* bat = (const float*)d_in[16];
  const float* Wv  = (const float*)d_in[17]; const float* bv  = (const float*)d_in[18];
  const float* Wo  = (const float*)d_in[19]; const float* bo  = (const float*)d_in[20];
  const float* g1  = (const float*)d_in[21]; const float* b1  = (const float*)d_in[22];
  const float* Wf1 = (const float*)d_in[23]; const float* bf1 = (const float*)d_in[24];
  const float* Wf2 = (const float*)d_in[25]; const float* bf2 = (const float*)d_in[26];
  const float* g2  = (const float*)d_in[27]; const float* b2  = (const float*)d_in[28];

  // workspace carve
  char* base = (char*)d_ws;
  size_t woff = 0;
  auto take = [&](size_t bytes) -> void* {
    void* p = base + woff;
    woff += (bytes + 255) & ~(size_t)255;
    return p;
  };
  const size_t NC = (size_t)NQ * CDIM;
  float*  xbuf    = (float*)take(NC * 4);
  bf16_t* xb      = (bf16_t*)take(NC * 2);
  float*  posbuf  = (float*)take(NC * 4);
  bf16_t* qb      = (bf16_t*)take(NC * 2);
  float*  vbuf    = (float*)take(NC * 4);
  bf16_t* msdab   = (bf16_t*)take(NC * 2);
  float*  tmpbuf  = (float*)take(NC * 4);
  float*  offbuf  = (float*)take((size_t)NQ * 384 * 4);
  float*  attbuf  = (float*)take((size_t)NQ * 128 * 4);
  bf16_t* hbuf    = (bf16_t*)take((size_t)NQ * DFFDIM * 2);
  float*  refbuf  = (float*)take((size_t)NQ * NLVL * 3 * 4);
  float*  vrbuf   = (float*)take(16 * 4);
  unsigned char* maskflat = (unsigned char*)take(NQ);
  const size_t PK_LAYER = 786432; // bf16 elems per layer (so,a,v,o,f1,f2)
  bf16_t* wpk     = (bf16_t*)take(NLAYER * PK_LAYER * 2);

  const int TB = 256;
  const int gridNC = (int)((NC + TB - 1) / TB);
  const int gridM  = (NQ + 63) / 64;   // 988 row-tiles of 64

  // --- one-time: pack inputs + weights, refs -------------------------------
  pack_src_kernel<<<gridNC, TB, 0, stream>>>(src[0], src[1], src[2], src[3],
                                             pos[0], pos[1], pos[2], pos[3],
                                             msk[0], msk[1], msk[2], msk[3],
                                             level_embed, xbuf, xb, posbuf, maskflat);
  valid_ratio_kernel<<<1, 32, 0, stream>>>(msk[0], msk[1], msk[2], msk[3], vrbuf);
  ref_points_kernel<<<(NQ * NLVL + TB - 1) / TB, TB, 0, stream>>>(vrbuf, refbuf);

  auto pack = [&](const float* W, bf16_t* dst, int K, int Nout) {
    int total = (Nout >> 4) * (K >> 5) * 32;
    pack_w_kernel<<<(total + TB - 1) / TB, TB, 0, stream>>>(W, dst, K, Nout);
  };
  for (int i = 0; i < NLAYER; ++i) {
    bf16_t* wl = wpk + (size_t)i * PK_LAYER;
    pack(Wso + (size_t)i * 256 * 384,  wl + 0,      256, 384);
    pack(Wat + (size_t)i * 256 * 128,  wl + 98304,  256, 128);
    pack(Wv  + (size_t)i * 256 * 256,  wl + 131072, 256, 256);
    pack(Wo  + (size_t)i * 256 * 256,  wl + 196608, 256, 256);
    pack(Wf1 + (size_t)i * 256 * 1024, wl + 262144, 256, 1024);
    pack(Wf2 + (size_t)i * 1024 * 256, wl + 524288, 1024, 256);
  }

  // --- encoder layers ------------------------------------------------------
  for (int i = 0; i < NLAYER; ++i) {
    bf16_t* wl = wpk + (size_t)i * PK_LAYER;
    // q = x + pos  (bf16, GEMM A operand)
    add_bf16_kernel<<<gridNC, TB, 0, stream>>>(xbuf, posbuf, qb, (int)NC);
    // sampling offsets / attention logits / values
    gemm_wmma_kernel<float><<<dim3(gridM, 384 / 64), 128, 0, stream>>>(
        qb, wl + 0, bso + (size_t)i * 384, offbuf, NQ, 256, 384, 0);
    gemm_wmma_kernel<float><<<dim3(gridM, 128 / 64), 128, 0, stream>>>(
        qb, wl + 98304, bat + (size_t)i * 128, attbuf, NQ, 256, 128, 0);
    softmax16_kernel<<<(NQ * NHEADS + TB - 1) / TB, TB, 0, stream>>>(attbuf, NQ * NHEADS);
    gemm_wmma_kernel<float><<<dim3(gridM, 256 / 64), 128, 0, stream>>>(
        xb, wl + 131072, bv + (size_t)i * 256, vbuf, NQ, 256, 256, 0);
    mask_v_kernel<<<gridNC, TB, 0, stream>>>(vbuf, maskflat);
    // deformable sampling (1 wave per query-head)
    msda_kernel<<<NQ, 256, 0, stream>>>(vbuf, offbuf, attbuf, refbuf, msdab);
    // output projection + residual LN
    gemm_wmma_kernel<float><<<dim3(gridM, 256 / 64), 128, 0, stream>>>(
        msdab, wl + 196608, bo + (size_t)i * 256, tmpbuf, NQ, 256, 256, 0);
    ln_res_kernel<<<(NQ + 7) / 8, 256, 0, stream>>>(xbuf, tmpbuf,
        g1 + (size_t)i * 256, b1 + (size_t)i * 256, xbuf, xb);
    // FFN (hidden kept in bf16 -> feeds next WMMA directly)
    gemm_wmma_kernel<bf16_t><<<dim3(gridM, 1024 / 64), 128, 0, stream>>>(
        xb, wl + 262144, bf1 + (size_t)i * 1024, hbuf, NQ, 256, 1024, 1);
    gemm_wmma_kernel<float><<<dim3(gridM, 256 / 64), 128, 0, stream>>>(
        hbuf, wl + 524288, bf2 + (size_t)i * 256, tmpbuf, NQ, 1024, 256, 0);
    ln_res_kernel<<<(NQ + 7) / 8, 256, 0, stream>>>(xbuf, tmpbuf,
        g2 + (size_t)i * 256, b2 + (size_t)i * 256, xbuf, xb);
  }

  // --- write output fmaps --------------------------------------------------
  unpack_kernel<<<gridNC, TB, 0, stream>>>(xbuf, (float*)d_out);
}
